// GeneralJEPAv2_57397942943884
// MI455X (gfx1250) — matrix operand order repaired and checked by reading, not compiled, not run on hardware
//
#include <hip/hip_runtime.h>
#include <hip/hip_bf16.h>

// ---------------------------------------------------------------------------
// CDNA5 (gfx1250, wave32) implementation of GeneralJEPAv2 forward.
// All 128-wide GEMMs use v_wmma_f32_16x16x32_f16 with fused bias + BN-stat
// reduction (LDS ds_add_f32 partials -> one global_atomic_add_f32 per column).
// Weights are pre-transposed/converted to f16 once per GEMM (k_wprep); each
// block stages them into LDS via the Tensor Data Mover (tensor_load_to_lds,
// with D# LDS-padding to keep a 136-half row pitch), falling back to a
// coalesced b128 copy loop when the TDM builtin is unavailable.
// Edge message/aggregation and segment means use f32 global atomics.
// ---------------------------------------------------------------------------

typedef __attribute__((ext_vector_type(16))) _Float16 v16h;
typedef __attribute__((ext_vector_type(8)))  float    v8f;
typedef __attribute__((ext_vector_type(4)))  unsigned int v4u;
typedef __attribute__((ext_vector_type(4)))  unsigned int u32x4;
typedef __attribute__((ext_vector_type(8)))  int      i32x8;
typedef __attribute__((ext_vector_type(4)))  int      i32x4;

union V16H {
    v16h     v;
    _Float16 h[16];
    v4u      q[2];
};

#define BN_EPS 1e-5f

__device__ __forceinline__ void atomAddG(float* p, float v) {
    __hip_atomic_fetch_add(p, v, __ATOMIC_RELAXED, __HIP_MEMORY_SCOPE_AGENT);
}
__device__ __forceinline__ void atomAddS(float* p, float v) {
    __hip_atomic_fetch_add(p, v, __ATOMIC_RELAXED, __HIP_MEMORY_SCOPE_WORKGROUP);
}

// ---------------------------------------------------------------------------
// zero fill
// ---------------------------------------------------------------------------
__global__ void k_zero_f(float* __restrict__ p, size_t n) {
    size_t i = (size_t)blockIdx.x * blockDim.x + threadIdx.x;
    if (i < n) p[i] = 0.0f;
}

// ---------------------------------------------------------------------------
// weight prep: W (f32, K x 128, k-major) -> Wt16 (f16, 128 x kpad, n-major,
// zero-padded K..kpad).  Tiny (16K elements); runs once per GEMM call.
// ---------------------------------------------------------------------------
__global__ void k_wprep(const float* __restrict__ W, _Float16* __restrict__ wt16,
                        int K, int kpad)
{
    int idx = blockIdx.x * blockDim.x + threadIdx.x;
    if (idx >= 128 * kpad) return;
    int n = idx / kpad;
    int k = idx - n * kpad;
    wt16[idx] = (k < K) ? (_Float16)W[(size_t)k * 128 + n] : (_Float16)0.0f;
}

// ---------------------------------------------------------------------------
// Z[rows x 128] = (X[rowIdx?][.] (+ X2[.])) @ W[K x 128] + bias
// Also accumulates per-column sum / sum-of-squares into stats[0:128]/[128:256]
// X row stride == K.  X2 (stride 128) only used when K == 128.
// One 256-thread block (8 wave32) computes a 128-row strip; each wave owns a
// 16-row strip and 8 accumulator tiles of v_wmma_f32_16x16x32_f16.
// ---------------------------------------------------------------------------
template<int K, bool HASX2>
__global__ __launch_bounds__(256) void k_gemm128_wmma(
    const float* __restrict__ X, const int* __restrict__ rowIdx,
    const float* __restrict__ X2,
    const _Float16* __restrict__ Wt16,  // 128 x kpad f16 (from k_wprep)
    const float* __restrict__ bias,     // 128 (may be null)
    float* __restrict__ Z,              // rows x 128
    float* __restrict__ stats,          // 256 floats (sum, sumsq) or null
    int rows)
{
    constexpr int kpad    = (K + 31) & ~31;
    constexpr int kblocks = kpad >> 5;

    __shared__ _Float16 Wt[128][136];   // transposed weights, f16, padded rows
    __shared__ float bsum[128];
    __shared__ float bsq[128];

    const int tid = threadIdx.x;

#if __has_builtin(__builtin_amdgcn_tensor_load_to_lds)
    // ---- TDM: one descriptor copies the whole 128 x kpad f16 tile into LDS,
    // inserting (136-kpad) halves of padding after every kpad-half row. ----
    if (tid < 32) {                     // wave 0 issues the tensor DMA
        const unsigned            lds0 = (unsigned)(unsigned long long)(&Wt[0][0]);
        const unsigned long long  ga   = (unsigned long long)Wt16;
        constexpr unsigned elems4 = (128u * (unsigned)kpad * 2u) / 4u; // DWORD count
        constexpr int rowDW  = kpad / 2;            // DWORDs per LDS row payload
        constexpr int padDW  = (136 - kpad) / 2;    // DWORDs of padding per row
        constexpr int piCode = (rowDW == 64) ? 5 : 3;  // pad_interval: 64 / 16 DW
        constexpr int paCode = padDW - 1;              // pad_amount: code = DW-1
        u32x4 g0;
        g0[0] = 1u;                                   // count=1 (valid descriptor)
        g0[1] = lds0;                                 // lds_addr
        g0[2] = (unsigned)ga;                         // global_addr[31:0]
        g0[3] = (unsigned)((ga >> 32) & 0x01ffffffull) | (2u << 30); // addr[56:32]|type=2
        i32x8 g1;
        g1[0] = (int)((2u << 16) | (1u << 20) |       // data_size=4B | pad_enable
                      ((unsigned)piCode << 22) | ((unsigned)paCode << 25));
        g1[1] = (int)((elems4 & 0xffffu) << 16);      // tensor_dim0[15:0]
        g1[2] = (int)(((elems4 >> 16) & 0xffffu) | (1u << 16)); // dim0[31:16]|dim1=1
        g1[3] = (int)((elems4 & 0xffffu) << 16);      // tile_dim0
        g1[4] = 0;                                    // tile_dim1/2 unused
        g1[5] = (int)elems4;                          // tensor_dim0_stride[31:0]
        g1[6] = 0;
        g1[7] = 0;
        i32x4 gz;  gz[0] = 0; gz[1] = 0; gz[2] = 0; gz[3] = 0;
        i32x8 gz8; gz8[0] = 0; gz8[1] = 0; gz8[2] = 0; gz8[3] = 0;
                   gz8[4] = 0; gz8[5] = 0; gz8[6] = 0; gz8[7] = 0;
        // clang-23 / amdgpu-toolchain 6-arg form:
        // (g0 u32x4, g1 i32x8, g2 i32x4, g3 i32x4, g4 i32x8, cpol i32)
        __builtin_amdgcn_tensor_load_to_lds(g0, g1, gz, gz, gz8, 0);
        __builtin_amdgcn_s_wait_tensorcnt(0);
    }
#else
    // ---- fallback: coalesced 16B-chunk copy into padded LDS rows ----
    {
        constexpr int CH = kpad / 8;                  // 16B chunks per row
        const v4u* gsrc = (const v4u*)Wt16;
        for (int c = tid; c < 128 * CH; c += 256) {
            int n  = c / CH;
            int cc = c - n * CH;
            *(v4u*)(&Wt[n][cc * 8]) = gsrc[c];
        }
    }
#endif
    if (tid < 128) { bsum[tid] = 0.0f; bsq[tid] = 0.0f; }
    __syncthreads();

    const int  wave    = tid >> 5;
    const int  lane    = tid & 31;
    const int  halfsel = lane >> 4;
    const int  ln      = lane & 15;
    const long rowBase = (long)blockIdx.x * 128 + (long)wave * 16;

    long srow = rowBase + ln;
    if (srow > (long)rows - 1) srow = (long)rows - 1;
    const long xrow = rowIdx ? (long)rowIdx[srow] : srow;
    const float* xr  = X + (size_t)xrow * (size_t)K;
    const float* x2r = HASX2 ? (X2 + (size_t)srow * 128) : nullptr;

    v8f acc[8] = {};

    #pragma unroll
    for (int kb = 0; kb < kblocks; ++kb) {
        // A fragment: lanes 0-15 rows M=0..15 hold K kb*32+{0..15};
        // lanes 16-31 hold K kb*32+{16..31}  (contiguous, matches B fragment)
        V16H af;
        #pragma unroll
        for (int h = 0; h < 2; ++h) {
            const int k0 = kb * 32 + halfsel * 16 + h * 8;
            // group is entirely in [0,K) or entirely out
            if (k0 + 8 <= K) {
                #pragma unroll
                for (int j = 0; j < 8; ++j) {
                    float v = xr[k0 + j];
                    if constexpr (HASX2) v += x2r[k0 + j];
                    af.h[h * 8 + j] = (_Float16)v;
                }
            } else {
                #pragma unroll
                for (int j = 0; j < 8; ++j) af.h[h * 8 + j] = (_Float16)0.0f;
            }
        }
        const int ks = kb * 32 + halfsel * 16;
        #pragma unroll
        for (int nt = 0; nt < 8; ++nt) {
            V16H bf;  // B fragment: lane = column, 16 consecutive K f16 (32B, aligned)
            const v4u* p = (const v4u*)(&Wt[nt * 16 + ln][ks]);
            bf.q[0] = p[0];
            bf.q[1] = p[1];
            acc[nt] = __builtin_amdgcn_wmma_f32_16x16x32_f16(
                false, af.v, false, bf.v, (short)0, acc[nt], false, false);
        }
    }

    // epilogue: bias add, store, per-column BN partial sums
    const bool full = ((long)(blockIdx.x + 1) * 128) <= (long)rows;
    if (full) {
        #pragma unroll
        for (int nt = 0; nt < 8; ++nt) {
            const int   n   = nt * 16 + ln;
            const float bn_ = bias ? bias[n] : 0.0f;
            float psum = 0.0f, psq = 0.0f;
            #pragma unroll
            for (int j = 0; j < 8; ++j) {
                const long r = rowBase + halfsel * 8 + j;  // C/D: VGPR j -> M = halfsel*8+j
                float v = acc[nt][j] + bn_;
                Z[(size_t)r * 128 + n] = v;
                psum += v; psq += v * v;
            }
            if (stats) { atomAddS(&bsum[n], psum); atomAddS(&bsq[n], psq); }
        }
    } else {
        #pragma unroll
        for (int nt = 0; nt < 8; ++nt) {
            const int   n   = nt * 16 + ln;
            const float bn_ = bias ? bias[n] : 0.0f;
            float psum = 0.0f, psq = 0.0f;
            #pragma unroll
            for (int j = 0; j < 8; ++j) {
                const long r = rowBase + halfsel * 8 + j;
                if (r < rows) {
                    float v = acc[nt][j] + bn_;
                    Z[(size_t)r * 128 + n] = v;
                    psum += v; psq += v * v;
                }
            }
            if (stats) { atomAddS(&bsum[n], psum); atomAddS(&bsq[n], psq); }
        }
    }
    __syncthreads();
    if (stats && tid < 128) {
        atomAddG(&stats[tid],       bsum[tid]);
        atomAddG(&stats[128 + tid], bsq[tid]);
    }
}

// ---------------------------------------------------------------------------
// BN finalize: scsh[0:128]=scale, scsh[128:256]=shift
// ---------------------------------------------------------------------------
__global__ void k_bn_finalize(const float* __restrict__ stats,
                              const float* __restrict__ g,
                              const float* __restrict__ be,
                              float* __restrict__ scsh, float M)
{
    int n = threadIdx.x;
    if (n < 128) {
        float mean = stats[n] / M;
        float var  = stats[128 + n] / M - mean * mean;
        float inv  = rsqrtf(var + BN_EPS);
        float sc   = g[n] * inv;
        scsh[n]       = sc;
        scsh[128 + n] = be[n] - mean * sc;
    }
}

// dst = (accumulate ? dst : 0) + relu(z*sc + sh)
__global__ void k_bn_apply(const float* __restrict__ z,
                           const float* __restrict__ scsh,
                           float* __restrict__ dst, int accumulate, size_t rows)
{
    size_t i = (size_t)blockIdx.x * blockDim.x + threadIdx.x;
    if (i >= rows * 128) return;
    int n = (int)(i & 127);
    float v = z[i] * scsh[n] + scsh[128 + n];
    v = v > 0.0f ? v : 0.0f;
    dst[i] = accumulate ? (dst[i] + v) : v;
}

// h0 = emb_node[x] + relu(bn(z))
__global__ void k_h0(const float* __restrict__ z, const float* __restrict__ scsh,
                     const float* __restrict__ emb_node, const int* __restrict__ x,
                     float* __restrict__ h0, size_t rows)
{
    size_t i = (size_t)blockIdx.x * blockDim.x + threadIdx.x;
    if (i >= rows * 128) return;
    size_t r = i >> 7;
    int    n = (int)(i & 127);
    float v = z[i] * scsh[n] + scsh[128 + n];
    v = v > 0.0f ? v : 0.0f;
    h0[i] = emb_node[(size_t)x[r] * 128 + n] + v;
}

// ---------------------------------------------------------------------------
// GINE message + scatter: aggr[dst] += relu(xn[src] + emb_edge[attr])
// one edge per 32 lanes, 4 columns per lane
// ---------------------------------------------------------------------------
__global__ void k_msg_aggr(const float* __restrict__ xn,
                           const int* __restrict__ src, const int* __restrict__ dst,
                           const int* __restrict__ emap, const int* __restrict__ eattr,
                           const float* __restrict__ embE,
                           float* __restrict__ aggr, long E)
{
    long idx = (long)blockIdx.x * blockDim.x + threadIdx.x;
    long e   = idx >> 5;
    if (e >= E) return;
    int lane = (int)(idx & 31);
    int s = src[e];
    int d = dst[e];
    int a = eattr[emap ? emap[e] : e];
    const float* xr = xn   + (size_t)s * 128;
    const float* er = embE + (size_t)a * 128;
    float*       ar = aggr + (size_t)d * 128;
    __builtin_prefetch(ar, 1, 1);
    int c0 = lane * 4;
    #pragma unroll
    for (int j = 0; j < 4; ++j) {
        int c = c0 + j;
        float m = xr[c] + er[c];
        m = m > 0.0f ? m : 0.0f;
        atomAddG(&ar[c], m);
    }
}

// ---------------------------------------------------------------------------
// segment sum:  sum[seg[r]] += v[rowmap?rowmap[r]:r];  cnt[seg[r]] += 1
// ---------------------------------------------------------------------------
__global__ void k_seg_sum(const float* __restrict__ v, const int* __restrict__ rowmap,
                          const int* __restrict__ seg,
                          float* __restrict__ sum, float* __restrict__ cnt, long rows)
{
    long idx = (long)blockIdx.x * blockDim.x + threadIdx.x;
    long r   = idx >> 5;
    if (r >= rows) return;
    int lane = (int)(idx & 31);
    int s = seg[r];
    const float* vr = v + (size_t)(rowmap ? rowmap[r] : (int)r) * 128;
    float* sr = sum + (size_t)s * 128;
    int c0 = lane * 4;
    #pragma unroll
    for (int j = 0; j < 4; ++j) atomAddG(&sr[c0 + j], vr[c0 + j]);
    if (lane == 0) atomAddG(&cnt[s], 1.0f);
}

// in-place: sum[i] /= max(cnt[i/128], 1)
__global__ void k_seg_fin(float* __restrict__ sum, const float* __restrict__ cnt, size_t segs)
{
    size_t i = (size_t)blockIdx.x * blockDim.x + threadIdx.x;
    if (i >= segs * 128) return;
    float c = cnt[i >> 7];
    sum[i] /= (c > 1.0f ? c : 1.0f);
}

// dst[r] = src[idx[r]] (rows of 128)
__global__ void k_gather128(const float* __restrict__ src, const int* __restrict__ idx,
                            float* __restrict__ dst, size_t rows)
{
    size_t i = (size_t)blockIdx.x * blockDim.x + threadIdx.x;
    if (i >= rows * 128) return;
    size_t r = i >> 7;
    dst[i] = src[(size_t)idx[r] * 128 + (i & 127)];
}

// exclusive cumsum of call_n_patches -> ctx/tgt flat subgraph indices
__global__ void k_make_idx(const int* __restrict__ np, const int* __restrict__ ctx,
                           const int* __restrict__ tgt,
                           int* __restrict__ ctxIdx, int* __restrict__ tgtIdx,
                           int G, int T)
{
    __shared__ int pre[1024];
    if (threadIdx.x == 0) {
        int acc = 0;
        for (int g = 0; g < G && g < 1024; ++g) { pre[g] = acc; acc += np[g]; }
    }
    __syncthreads();
    for (int g = threadIdx.x; g < G; g += blockDim.x) {
        ctxIdx[g] = ctx[g] + pre[g];
        for (int t = 0; t < T; ++t) tgtIdx[g * T + t] = tgt[g * T + t] + pre[g];
    }
}

// cond[g*T+t] = visctx[g] + pes[g*T+t]
__global__ void k_cond_add(const float* __restrict__ visctx, const float* __restrict__ pes,
                           float* __restrict__ cond, size_t GT, int T)
{
    size_t i = (size_t)blockIdx.x * blockDim.x + threadIdx.x;
    if (i >= GT * 128) return;
    size_t r = i >> 7;
    size_t g = r / (size_t)T;
    cond[i] = visctx[g * 128 + (i & 127)] + pes[i];
}

// vis_target[g] = embedded_target[g, 0, :]
__global__ void k_vis_tgt(const float* __restrict__ tgt, float* __restrict__ vis,
                          size_t G, int T)
{
    size_t i = (size_t)blockIdx.x * blockDim.x + threadIdx.x;
    if (i >= G * 128) return;
    size_t g = i >> 7;
    vis[i] = tgt[g * (size_t)T * 128 + (i & 127)];
}

// ---------------------------------------------------------------------------
// host-side orchestration
// ---------------------------------------------------------------------------
struct Lin { const float *W, *b, *g, *be; };

static inline unsigned int nblk(size_t n, int b) { return (unsigned int)((n + b - 1) / b); }

extern "C" void kernel_launch(void* const* d_in, const int* in_sizes, int n_in,
                              void* d_out, int out_size, void* d_ws, size_t ws_size,
                              hipStream_t stream)
{
    (void)n_in; (void)out_size; (void)ws_size;

    const int*   x        = (const int*)d_in[0];
    const int*   edgeattr = (const int*)d_in[1];
    const float* rw       = (const float*)d_in[2];
    const int*   nmap     = (const int*)d_in[3];   // subgraphs_nodes_mapper (NS)
    const int*   csub     = (const int*)d_in[4];   // combined_subgraphs (2,ES)
    const int*   emap     = (const int*)d_in[5];   // subgraphs_edges_mapper (ES)
    const int*   sbatch   = (const int*)d_in[6];   // subgraphs_batch (NS)
    const int*   eidx     = (const int*)d_in[7];   // edge_index (2,E)
    const int*   batch    = (const int*)d_in[8];   // (N)
    const int*   npatch   = (const int*)d_in[9];   // (G)
    const int*   ctxsub   = (const int*)d_in[10];  // (G)
    const int*   tgtsub   = (const int*)d_in[11];  // (G,T)
    const float* patch_pe = (const float*)d_in[12];// (P,16)

    const int N  = in_sizes[0];
    const int E  = in_sizes[1];
    const int NS = in_sizes[3];
    const int ES = in_sizes[5];
    const int G  = in_sizes[9];
    const int T  = in_sizes[11] / G;
    const int P  = in_sizes[12] / 16;
    const int GT = G * T;

    const float* emb_node = (const float*)d_in[13];
    const float* emb_edge = (const float*)d_in[14];
    auto LIN = [&](int base) {
        Lin l; l.W = (const float*)d_in[base]; l.b = (const float*)d_in[base + 1];
        l.g = (const float*)d_in[base + 2]; l.be = (const float*)d_in[base + 3]; return l;
    };
    Lin rw_enc   = LIN(15);
    Lin patch_rw = LIN(19);
    Lin ctx_gnn[3] = { LIN(23), LIN(27), LIN(31) };
    Lin ctx_u[2]   = { LIN(35), LIN(39) };
    Lin tgt_gnn[3] = { LIN(43), LIN(47), LIN(51) };
    Lin tgt_u[2]   = { LIN(55), LIN(59) };
    Lin pl1 = LIN(63), pl2 = LIN(67);
    Lin pl3; pl3.W = (const float*)d_in[71]; pl3.b = (const float*)d_in[72];
    pl3.g = nullptr; pl3.be = nullptr;

    // ---- workspace carve (floats) ----
    const size_t maxR = (size_t)(NS > N ? NS : N);
    float* ws = (float*)d_ws;
    float* z      = ws;               ws += maxR * 128;   // GEMM outputs
    float* bufA   = ws;               ws += maxR * 128;   // aggr / node seg-mean
    float* h0     = ws;               ws += (size_t)N * 128;
    float* xs     = ws;               ws += (size_t)NS * 128;
    float* xt     = ws;               ws += (size_t)N * 128;
    float* segP   = ws;               ws += (size_t)P * 128;
    float* embSub = ws;               ws += (size_t)P * 128;
    float* subFull= ws;               ws += (size_t)P * 128;
    float* segG   = ws;               ws += (size_t)G * 128;
    float* cntP   = ws;               ws += P;
    float* cntN   = ws;               ws += N;
    float* cntG   = ws;               ws += G;
    float* stats  = ws;               ws += 256;
    float* scsh   = ws;               ws += 256;
    _Float16* wt16 = (_Float16*)ws;   ws += 8192;         // 128x128 f16 staging
    float* pes    = ws;               ws += (size_t)GT * 128;
    float* cond   = ws;               ws += (size_t)GT * 128;
    float* hb     = ws;               ws += (size_t)GT * 128;
    int*   ctxIdx = (int*)ws;
    int*   tgtIdx = ctxIdx + G;

    // ---- output layout ----
    float* out          = (float*)d_out;
    float* out_target   = out;                               // (G,T,128)
    float* out_pred     = out + (size_t)GT * 128;            // (G,T,128)
    float* out_visctx   = out + (size_t)2 * GT * 128;        // (G,128)
    float* out_vistgt   = out_visctx + (size_t)G * 128;      // (G,128)
    float* out_visgraph = out_vistgt + (size_t)G * 128;      // (G,128)

    auto ZERO = [&](float* p, size_t n) {
        k_zero_f<<<nblk(n, 256), 256, 0, stream>>>(p, n);
    };
    // weight prep + GEMM + (optional) BN-stat finalize
    auto GEMM = [&](const float* X, const int* ridx, const float* X2, const Lin& L,
                    float* Zo, int rows, int K, bool doStats) {
        const int kpad = (K + 31) & ~31;
        k_wprep<<<nblk((size_t)128 * kpad, 256), 256, 0, stream>>>(L.W, wt16, K, kpad);
        if (doStats) ZERO(stats, 256);
        float* st = doStats ? stats : nullptr;
        unsigned int grid = (unsigned int)((rows + 127) / 128);
        if (K == 16) {
            k_gemm128_wmma<16, false><<<grid, 256, 0, stream>>>(
                X, ridx, nullptr, wt16, L.b, Zo, st, rows);
        } else if (X2) {
            k_gemm128_wmma<128, true><<<grid, 256, 0, stream>>>(
                X, ridx, X2, wt16, L.b, Zo, st, rows);
        } else {
            k_gemm128_wmma<128, false><<<grid, 256, 0, stream>>>(
                X, ridx, nullptr, wt16, L.b, Zo, st, rows);
        }
        if (doStats)
            k_bn_finalize<<<1, 128, 0, stream>>>(stats, L.g, L.be, scsh, (float)rows);
    };
    auto APPLY = [&](const float* zz, float* dst, int accum, size_t rows) {
        k_bn_apply<<<nblk(rows * 128, 256), 256, 0, stream>>>(zz, scsh, dst, accum, rows);
    };
    auto SEGSUM = [&](const float* v, const int* rmap, const int* seg,
                      float* sum, float* cnt, long rows, size_t segs) {
        ZERO(sum, segs * 128);
        ZERO(cnt, segs);
        k_seg_sum<<<nblk((size_t)rows * 32, 256), 256, 0, stream>>>(v, rmap, seg, sum, cnt, rows);
        k_seg_fin<<<nblk(segs * 128, 256), 256, 0, stream>>>(sum, cnt, segs);
    };

    // ================= h0 = emb_node[x] + mlp(rw_pos_enc) =================
    GEMM(rw, nullptr, nullptr, rw_enc, z, N, 16, true);
    k_h0<<<nblk((size_t)N * 128, 256), 256, 0, stream>>>(z, scsh, emb_node, x, h0, (size_t)N);

    // ================= context branch: xs over NS subgraph nodes ==========
    k_gather128<<<nblk((size_t)NS * 128, 256), 256, 0, stream>>>(h0, nmap, xs, (size_t)NS);

    for (int i = 0; i < 3; ++i) {
        if (i > 0) {
            // xs += mlp(seg_mean(xs, sbatch, P)[sbatch])
            SEGSUM(xs, nullptr, sbatch, segP, cntP, NS, (size_t)P);
            GEMM(segP, sbatch, nullptr, ctx_u[i - 1], z, NS, 128, true);
            APPLY(z, xs, 1, (size_t)NS);
            // xs = seg_mean(xs, nmap, N)[nmap]
            SEGSUM(xs, nullptr, nmap, bufA, cntN, NS, (size_t)N);
            k_gather128<<<nblk((size_t)NS * 128, 256), 256, 0, stream>>>(bufA, nmap, xs, (size_t)NS);
        }
        // GINE layer
        ZERO(bufA, (size_t)NS * 128);
        k_msg_aggr<<<nblk((size_t)ES * 32, 256), 256, 0, stream>>>(
            xs, csub, csub + ES, emap, edgeattr, emb_edge, bufA, (long)ES);
        GEMM(xs, nullptr, bufA, ctx_gnn[i], z, NS, 128, true);
        APPLY(z, xs, 1, (size_t)NS);
    }
    // embedded_subgraph = seg_mean(xs, sbatch, P)
    SEGSUM(xs, nullptr, sbatch, embSub, cntP, NS, (size_t)P);

    // flat patch indices
    k_make_idx<<<1, 256, 0, stream>>>(npatch, ctxsub, tgtsub, ctxIdx, tgtIdx, G, T);
    // embedded_context / vis_context
    k_gather128<<<nblk((size_t)G * 128, 256), 256, 0, stream>>>(embSub, ctxIdx, out_visctx, (size_t)G);

    // ================= target branch: xt over N full-graph nodes ==========
    (void)hipMemcpyAsync(xt, h0, (size_t)N * 128 * sizeof(float),
                         hipMemcpyDeviceToDevice, stream);
    for (int i = 0; i < 3; ++i) {
        if (i > 0) {
            SEGSUM(xt, nullptr, batch, segG, cntG, N, (size_t)G);
            GEMM(segG, batch, nullptr, tgt_u[i - 1], z, N, 128, true);
            APPLY(z, xt, 1, (size_t)N);
        }
        ZERO(bufA, (size_t)N * 128);
        k_msg_aggr<<<nblk((size_t)E * 32, 256), 256, 0, stream>>>(
            xt, eidx, eidx + E, nullptr, edgeattr, emb_edge, bufA, (long)E);
        GEMM(xt, nullptr, bufA, tgt_gnn[i], z, N, 128, true);
        APPLY(z, xt, 1, (size_t)N);
    }

    // vis_graph = seg_mean(xt, batch, G) (directly into output region)
    SEGSUM(xt, nullptr, batch, out_visgraph, cntG, N, (size_t)G);

    // sub_from_full = seg_mean(xt[nmap], sbatch, P)
    SEGSUM(xt, nmap, sbatch, subFull, cntP, NS, (size_t)P);

    // embedded_target = sub_from_full[tgtIdx]; vis_target = [:,0,:]
    k_gather128<<<nblk((size_t)GT * 128, 256), 256, 0, stream>>>(subFull, tgtIdx, out_target, (size_t)GT);
    k_vis_tgt<<<nblk((size_t)G * 128, 256), 256, 0, stream>>>(out_target, out_vistgt, (size_t)G, T);

    // ================= predictor head ====================================
    // pes = mlp(patch_pe[tgtIdx])      (K=16)
    GEMM(patch_pe, tgtIdx, nullptr, patch_rw, hb, GT, 16, true);
    APPLY(hb, pes, 0, (size_t)GT);
    // cond = embedded_context[:,None,:] + pes
    k_cond_add<<<nblk((size_t)GT * 128, 256), 256, 0, stream>>>(out_visctx, pes, cond, (size_t)GT, T);
    // h = relu(bn(cond @ l1))
    GEMM(cond, nullptr, nullptr, pl1, hb, GT, 128, true);
    APPLY(hb, pes, 0, (size_t)GT);
    // h = relu(bn(h @ l2))
    GEMM(pes, nullptr, nullptr, pl2, hb, GT, 128, true);
    APPLY(hb, cond, 0, (size_t)GT);
    // pred = h @ l3 + b  (no BN)
    GEMM(cond, nullptr, nullptr, pl3, out_pred, GT, 128, false);
}
